// hGRU_80668075753563
// MI455X (gfx1250) — matrix operand order repaired and checked by reading, not compiled, not run on hardware
//
#include <hip/hip_runtime.h>
#include <hip/hip_bf16.h>

// ---------------- problem constants ----------------
#define BATCH   16
#define C       25
#define CP      32          // channel dim padded to 32 (WMMA K / N tile)
#define H       150
#define W       150
#define HP      164         // H + 2*7 halo
#define WP      176         // >= W + 7 + 7 + 9 (edge-tile A overreach), 16B friendly
#define NPIX    (BATCH * H * W)   // 360000
#define TSTEPS  8
#define RBLK    64

typedef __attribute__((ext_vector_type(16))) __bf16 v16bf;
typedef __attribute__((ext_vector_type(8)))  __bf16 v8bf;
typedef __attribute__((ext_vector_type(8)))  float  v8f;

typedef __hip_bfloat16 bf16;

__device__ __forceinline__ float bf2f(bf16 v) { return __bfloat162float(v); }
__device__ __forceinline__ bf16  f2bf(float v) { return __float2bfloat16(v); }

// flat pixel id -> element index in zero-halo padded layout [B][HP][WP][CP]
__device__ __forceinline__ size_t padidx(int pix, int c) {
  int b   = pix / (H * W);
  int rem = pix - b * (H * W);
  int y   = rem / W;
  int x   = rem - y * W;
  return (((size_t)(b * HP + y + 7)) * WP + (size_t)(x + 7)) * CP + c;
}

// A-fragment (16x32 bf16, wave32): lane L (L<16) = row M=L holds K 0..7,16..23;
// lane L (>=16) = row M=L-16 holds K 8..15,24..31. kbase = (lane>=16)?8:0.
__device__ __forceinline__ v16bf load_a16(const bf16* p /* pixel base + kbase */) {
  const v8bf* q = (const v8bf*)p;
  v8bf lo = q[0];          // K kbase .. kbase+7
  v8bf hi = q[2];          // K kbase+16 .. kbase+23
  return __builtin_shufflevector(lo, hi, 0, 1, 2, 3, 4, 5, 6, 7,
                                         8, 9, 10, 11, 12, 13, 14, 15);
}

#define WMMA_BF16(a, b, c) \
  __builtin_amdgcn_wmma_f32_16x16x32_bf16(false, (a), false, (b), (short)0, (c), false, false)

// ---------------- weight packing ----------------
// B-fragment register image: lane L holds column N=L&15, K = (e&7) + 8*(L>=16) + 16*(e>=8),
// 16 bf16 contiguous per lane. Layout: [kykx][ntile][lane][16]
__global__ void pack_w15(const float* __restrict__ wsrc, bf16* __restrict__ dst) {
  int idx = blockIdx.x * 256 + threadIdx.x;            // 225*2*32*16 = 230400
  if (idx >= 225 * 1024) return;
  int e    = idx & 15;
  int lane = (idx >> 4) & 31;
  int nt   = (idx >> 9) & 1;
  int kykx = idx >> 10;
  int K = (e & 7) + ((lane >= 16) ? 8 : 0) + ((e >= 8) ? 16 : 0); // input channel
  int N = (lane & 15) + nt * 16;                       // output channel
  int ky = kykx / 15, kx = kykx % 15;
  float v = (K < C && N < C) ? wsrc[((N * C + K) * 15 + ky) * 15 + kx] : 0.f;
  dst[idx] = f2bf(v);
}

__global__ void pack_w1(const float* __restrict__ wsrc, bf16* __restrict__ dst) {
  int idx = blockIdx.x * 256 + threadIdx.x;            // 2*32*16 = 1024
  if (idx >= 1024) return;
  int e    = idx & 15;
  int lane = (idx >> 4) & 31;
  int nt   = idx >> 9;
  int K = (e & 7) + ((lane >= 16) ? 8 : 0) + ((e >= 8) ? 16 : 0);
  int N = (lane & 15) + nt * 16;
  float v = (K < C && N < C) ? wsrc[N * C + K] : 0.f;
  dst[idx] = f2bf(v);
}

// ---------------- 15x15 conv, implicit GEMM via WMMA ----------------
// One wave = 32 pixels x 32 out-channels: 2 A tiles x 2 B tiles ->
// 4 independent acc chains, 4 WMMAs per tap, B traffic halved per FLOP.
// Input is zero-halo padded -> every load unconditional, immediate offsets.
__global__ void __launch_bounds__(256)
conv15_wmma(const bf16* __restrict__ in, const bf16* __restrict__ wpack,
            bf16* __restrict__ out) {
  const int lane = threadIdx.x & 31;
  int tile = blockIdx.x * 8 + (threadIdx.x >> 5);      // 16*150*5 = 12000 tiles
  const int xt = tile % 5;  tile /= 5;
  const int y  = tile % H;
  const int b  = tile / H;
  const int x0 = xt * 32;
  const int l15 = lane & 15;
  const int kbase = (lane >= 16) ? 8 : 0;

  // A base for (ky=0,kx=0): padded coords (y-7+7, xo-7+7) = (y, x0+l15)
  const bf16* arow = in + (((size_t)(b * HP + y)) * WP + (size_t)(x0 + l15)) * CP + kbase;
  const bf16* wl = wpack + lane * 16;                  // + kykx*1024 (+512 for nt1)

  v8f acc00 = {}, acc01 = {}, acc10 = {}, acc11 = {};
#pragma unroll 1
  for (int ky = 0; ky < 15; ++ky) {
#pragma unroll
    for (int kx = 0; kx < 15; ++kx) {
      v16bf a0 = load_a16(arow + kx * CP);             // pixels x0   .. x0+15
      v16bf a1 = load_a16(arow + (kx + 16) * CP);      // pixels x0+16.. x0+31
      const int kk = ky * 15 + kx;
      v16bf b0 = *(const v16bf*)(wl + (size_t)kk * 1024);
      v16bf b1 = *(const v16bf*)(wl + (size_t)kk * 1024 + 512);
      acc00 = WMMA_BF16(a0, b0, acc00);
      acc01 = WMMA_BF16(a0, b1, acc01);
      acc10 = WMMA_BF16(a1, b0, acc10);
      acc11 = WMMA_BF16(a1, b1, acc11);
    }
    arow += (size_t)WP * CP;
  }
  // C/D layout: VGPR r -> M = r + 8*(lane>=16), N = lane&15
  const int mo = (lane >= 16) ? 8 : 0;
  bf16* orow = out + ((size_t)(b * H + y) * W) * CP;   // raw output: flat layout
#pragma unroll
  for (int r = 0; r < 8; ++r) {
    const int px0 = x0 + mo + r;
    const int px1 = px0 + 16;
    if (px0 < W) {
      orow[(size_t)px0 * CP + l15]      = f2bf(acc00[r]);
      orow[(size_t)px0 * CP + 16 + l15] = f2bf(acc01[r]);
    }
    if (px1 < W) {
      orow[(size_t)px1 * CP + l15]      = f2bf(acc10[r]);
      orow[(size_t)px1 * CP + 16 + l15] = f2bf(acc11[r]);
    }
  }
}

// ---------------- 1x1 conv via WMMA (padded in, flat out) -------------------
__global__ void __launch_bounds__(256)
conv1_wmma(const bf16* __restrict__ in, const bf16* __restrict__ wpack,
           bf16* __restrict__ out) {
  const int lane = threadIdx.x & 31;
  const int mt = blockIdx.x * 8 + (threadIdx.x >> 5);  // 22500 M-tiles
  if (mt >= NPIX / 16) return;                         // wave-uniform guard
  const int l15 = lane & 15;
  const int kbase = (lane >= 16) ? 8 : 0;
  const int pix = mt * 16 + l15;

  v16bf a  = load_a16(in + padidx(pix, kbase));
  v16bf b0 = *(const v16bf*)(wpack + lane * 16);
  v16bf b1 = *(const v16bf*)(wpack + 512 + lane * 16);
  v8f acc0 = {}, acc1 = {};
  acc0 = WMMA_BF16(a, b0, acc0);
  acc1 = WMMA_BF16(a, b1, acc1);
  const int mo = (lane >= 16) ? 8 : 0;
  bf16* op = out + (size_t)mt * 16 * CP;
#pragma unroll
  for (int r = 0; r < 8; ++r) {
    op[(size_t)(mo + r) * CP + l15]      = f2bf(acc0[r]);
    op[(size_t)(mo + r) * CP + 16 + l15] = f2bf(acc1[r]);
  }
}

// ---------------- BN statistics (sum, sumsq per channel) --------------------
__global__ void __launch_bounds__(256)
stats32(const bf16* __restrict__ in, float* __restrict__ sums, int padded) {
  __shared__ float s1[256], s2[256];
  const int c = threadIdx.x & 31, slot = threadIdx.x >> 5;
  float a = 0.f, b = 0.f;
  for (int pix = blockIdx.x * 8 + slot; pix < NPIX; pix += gridDim.x * 8) {
    size_t ad = padded ? padidx(pix, c) : ((size_t)pix * CP + c);
    float v = bf2f(in[ad]);
    a += v; b += v * v;
  }
  s1[threadIdx.x] = a; s2[threadIdx.x] = b;
  __syncthreads();
  if (slot == 0) {
#pragma unroll
    for (int k = 1; k < 8; ++k) { a += s1[k * 32 + c]; b += s2[k * 32 + c]; }
    atomicAdd(&sums[c], a);
    atomicAdd(&sums[32 + c], b);
  }
}

__global__ void bn_fin(const float* __restrict__ sums, const float* __restrict__ g,
                       const float* __restrict__ bb, float eps, float* __restrict__ par) {
  int c = threadIdx.x;
  if (c >= 32) return;
  const float inv_n = 1.f / (float)NPIX;
  float m = sums[c] * inv_n;
  float v = sums[32 + c] * inv_n - m * m;
  float gg  = (c < C) ? g[c]  : 0.f;
  float bbv = (c < C) ? bb[c] : 0.f;
  float s = gg * rsqrtf(v + eps);
  par[c] = s;
  par[32 + c] = bbv - m * s;
}

// ---------------- fused elementwise kernels ----------------
__global__ void __launch_bounds__(256)
feat_k(const float* __restrict__ x, const float* __restrict__ fw,
       const float* __restrict__ fb, bf16* __restrict__ xf) {
  const size_t idx = (size_t)blockIdx.x * 256 + threadIdx.x;   // NPIX*32
  const int c = (int)(idx & 31);
  const int pix = (int)(idx >> 5);
  const int b = pix / (H * W);
  const int rem = pix - b * (H * W);
  const int y = rem / W, xx = rem - (rem / W) * W;
  float acc = 0.f;
  if (c < C) {
    acc = fb[c];
    const float* xim = x + (size_t)b * H * W;
    const float* wc  = fw + c * 49;
#pragma unroll
    for (int ky = 0; ky < 7; ++ky) {
      int yi = y + ky - 3;
      if ((unsigned)yi >= (unsigned)H) continue;
#pragma unroll
      for (int kx = 0; kx < 7; ++kx) {
        int xi = xx + kx - 3;
        if ((unsigned)xi < (unsigned)W) acc += xim[yi * W + xi] * wc[ky * 7 + kx];
      }
    }
    acc = acc * acc;                                   // squaring nonlinearity
  }
  xf[idx] = f2bf(acc);
}

// g1 = sigmoid(bn(u1)); hg = h2*g1   (u1 flat, h2/hg padded)
__global__ void __launch_bounds__(256)
gate1_k(const bf16* __restrict__ u1, const float* __restrict__ par,
        const bf16* __restrict__ h2, bf16* __restrict__ hg) {
  const size_t idx = (size_t)blockIdx.x * 256 + threadIdx.x;
  const int c = (int)(idx & 31);
  const int pix = (int)(idx >> 5);
  const size_t pa = padidx(pix, c);
  float z  = bf2f(u1[idx]) * par[c] + par[32 + c];
  float g1 = 1.f / (1.f + expf(-z));
  float v  = bf2f(h2[pa]) * g1;
  hg[pa] = f2bf(c < C ? v : 0.f);
}

// h1 = tanh(xf - bn(c1)*(alpha*h2 + mu))   (c1/xf flat, h2/h1 padded)
__global__ void __launch_bounds__(256)
h1_k(const bf16* __restrict__ c1r, const float* __restrict__ par,
     const bf16* __restrict__ xf, const bf16* __restrict__ h2,
     const float* __restrict__ alpha, const float* __restrict__ mu,
     bf16* __restrict__ h1) {
  const size_t idx = (size_t)blockIdx.x * 256 + threadIdx.x;
  const int c = (int)(idx & 31);
  const int pix = (int)(idx >> 5);
  const size_t pa = padidx(pix, c);
  float c1  = bf2f(c1r[idx]) * par[c] + par[32 + c];
  float h2v = bf2f(h2[pa]);
  float av  = (c < C) ? alpha[c] : 0.f;
  float mv  = (c < C) ? mu[c]    : 0.f;
  float v   = tanhf(bf2f(xf[idx]) - c1 * (av * h2v + mv));
  h1[pa] = f2bf(c < C ? v : 0.f);
}

// h2 = (1-g2)*h2 + g2*tanh(kappa*h1 + gamma*c2 + omega*h1*c2)
// (u2/c2 flat, h1/h2 padded)
__global__ void __launch_bounds__(256)
combine_k(const bf16* __restrict__ u2, const float* __restrict__ pA,
          const bf16* __restrict__ c2, const float* __restrict__ pB,
          const bf16* __restrict__ h1, bf16* __restrict__ h2,
          const float* __restrict__ kappa, const float* __restrict__ gam,
          const float* __restrict__ omega) {
  const size_t idx = (size_t)blockIdx.x * 256 + threadIdx.x;
  const int c = (int)(idx & 31);
  const int pix = (int)(idx >> 5);
  const size_t pa = padidx(pix, c);
  float g2  = 1.f / (1.f + expf(-(bf2f(u2[idx]) * pA[c] + pA[32 + c])));
  float c2v = bf2f(c2[idx]) * pB[c] + pB[32 + c];
  float h1v = bf2f(h1[pa]);
  float h2v = bf2f(h2[pa]);
  float kc = (c < C) ? kappa[c] : 0.f;
  float gc = (c < C) ? gam[c]   : 0.f;
  float oc = (c < C) ? omega[c] : 0.f;
  float h2t = tanhf(kc * h1v + gc * c2v + oc * h1v * c2v);
  float r = (1.f - g2) * h2v + g2 * h2t;
  h2[pa] = f2bf(c < C ? r : 0.f);
}

// ---------------- readout ----------------
__global__ void __launch_bounds__(256)
readout_max_k(const bf16* __restrict__ h2, const float* __restrict__ par,
              const float* __restrict__ w6, float* __restrict__ partial) {
  const int b   = blockIdx.x / RBLK;
  const int blk = blockIdx.x % RBLK;
  __shared__ float r0[256], r1[256];
  float m0 = -3.0e38f, m1 = -3.0e38f;
  for (int p = blk * 256 + threadIdx.x; p < H * W; p += RBLK * 256) {
    const bf16* pp = h2 + padidx(b * (H * W) + p, 0);
    float s0 = 0.f, s1 = 0.f;
#pragma unroll
    for (int c = 0; c < C; ++c) {
      float v = bf2f(pp[c]) * par[c] + par[32 + c];
      s0 += v * w6[c];
      s1 += v * w6[C + c];
    }
    m0 = fmaxf(m0, s0);
    m1 = fmaxf(m1, s1);
  }
  r0[threadIdx.x] = m0; r1[threadIdx.x] = m1;
  __syncthreads();
  for (int s = 128; s > 0; s >>= 1) {
    if (threadIdx.x < s) {
      r0[threadIdx.x] = fmaxf(r0[threadIdx.x], r0[threadIdx.x + s]);
      r1[threadIdx.x] = fmaxf(r1[threadIdx.x], r1[threadIdx.x + s]);
    }
    __syncthreads();
  }
  if (threadIdx.x == 0) {
    partial[((size_t)b * RBLK + blk) * 2 + 0] = r0[0];
    partial[((size_t)b * RBLK + blk) * 2 + 1] = r1[0];
  }
}

__global__ void final_k(const float* __restrict__ partial, const float* __restrict__ bn1_g,
                        const float* __restrict__ bn1_b, const float* __restrict__ fc_w,
                        const float* __restrict__ fc_b, float* __restrict__ out) {
  __shared__ float y[16][2];
  __shared__ float yn[16][2];
  const int t = threadIdx.x;
  if (t < 32) {
    int b = t >> 1, o = t & 1;
    float m = -3.0e38f;
    for (int i = 0; i < RBLK; ++i)
      m = fmaxf(m, partial[((size_t)b * RBLK + i) * 2 + o]);
    y[b][o] = m;
  }
  __syncthreads();
  if (t < 2) {
    float s = 0.f, s2 = 0.f;
    for (int b = 0; b < 16; ++b) { s += y[b][t]; s2 += y[b][t] * y[b][t]; }
    float m = s * (1.f / 16.f);
    float v = s2 * (1.f / 16.f) - m * m;
    float sc = rsqrtf(v + 1e-5f) * bn1_g[t];
    for (int b = 0; b < 16; ++b) yn[b][t] = (y[b][t] - m) * sc + bn1_b[t];
  }
  __syncthreads();
  if (t < 32) {
    int b = t >> 1, k = t & 1;
    out[b * 2 + k] = yn[b][0] * fc_w[k * 2 + 0] + yn[b][1] * fc_w[k * 2 + 1] + fc_b[k];
  }
}

// ---------------- host orchestration ----------------
extern "C" void kernel_launch(void* const* d_in, const int* in_sizes, int n_in,
                              void* d_out, int out_size, void* d_ws, size_t ws_size,
                              hipStream_t stream) {
  (void)in_sizes; (void)n_in; (void)out_size; (void)ws_size;
  const float* x       = (const float*)d_in[0];
  const float* feat_w  = (const float*)d_in[1];
  const float* feat_b  = (const float*)d_in[2];
  const float* gain_w  = (const float*)d_in[3];
  const float* mix_w   = (const float*)d_in[4];
  const float* w_inh   = (const float*)d_in[5];
  const float* w_exc   = (const float*)d_in[6];
  const float* alpha   = (const float*)d_in[7];
  const float* gamma_p = (const float*)d_in[8];
  const float* kappa   = (const float*)d_in[9];
  const float* omega   = (const float*)d_in[10];
  const float* mu      = (const float*)d_in[11];
  const float* bn_g    = (const float*)d_in[12];
  const float* bn_b    = (const float*)d_in[13];
  const float* bn2_g   = (const float*)d_in[14];
  const float* bn2_b   = (const float*)d_in[15];
  const float* conv6_w = (const float*)d_in[16];
  const float* bn1_g   = (const float*)d_in[17];
  const float* bn1_b   = (const float*)d_in[18];
  const float* fc_w    = (const float*)d_in[19];
  const float* fc_b    = (const float*)d_in[20];
  float* out = (float*)d_out;

  // workspace carving (all re-initialized every call; ~160 MB total)
  char* wptr = (char*)d_ws;
  auto carve = [&](size_t bytes) {
    char* p = wptr;
    wptr += (bytes + 255) & ~(size_t)255;
    return p;
  };
  const size_t padBytes  = (size_t)BATCH * HP * WP * CP * sizeof(bf16); // 29.6 MB
  const size_t flatBytes = (size_t)(NPIX + 64) * CP * sizeof(bf16);     // 23.0 MB
  bf16* h2b = (bf16*)carve(padBytes);   // padded (conv input / recurrent state)
  bf16* h1b = (bf16*)carve(padBytes);   // padded (conv input)
  bf16* hgb = (bf16*)carve(padBytes);   // padded (conv input)
  bf16* xfb = (bf16*)carve(flatBytes);  // flat (elementwise only)
  bf16* ur  = (bf16*)carve(flatBytes);  // flat raw conv out (u1 / u2)
  bf16* cr  = (bf16*)carve(flatBytes);  // flat raw conv out (c1 / c2)
  bf16* wpI = (bf16*)carve((size_t)225 * 1024 * sizeof(bf16));
  bf16* wpE = (bf16*)carve((size_t)225 * 1024 * sizeof(bf16));
  bf16* wpG = (bf16*)carve((size_t)1024 * sizeof(bf16));
  bf16* wpM = (bf16*)carve((size_t)1024 * sizeof(bf16));
  float* sums    = (float*)carve(64 * sizeof(float));
  float* bnA     = (float*)carve(64 * sizeof(float));
  float* bnB     = (float*)carve(64 * sizeof(float));
  float* partial = (float*)carve((size_t)BATCH * RBLK * 2 * sizeof(float));

  const int EW_BLOCKS  = NPIX * CP / 256;          // 45000
  const int C15_BLOCKS = BATCH * H * 5 / 8;        // 1500 (12000 wave tiles)
  const int C1_BLOCKS  = (NPIX / 16 + 7) / 8;      // 2813 (22500 wave tiles)

  // zero-halo init (h2 also serves as h2(t=0) = 0) + weight packing + features
  (void)hipMemsetAsync(h2b, 0, padBytes, stream);
  (void)hipMemsetAsync(h1b, 0, padBytes, stream);
  (void)hipMemsetAsync(hgb, 0, padBytes, stream);
  pack_w15<<<900, 256, 0, stream>>>(w_inh, wpI);
  pack_w15<<<900, 256, 0, stream>>>(w_exc, wpE);
  pack_w1 <<<4, 256, 0, stream>>>(gain_w, wpG);
  pack_w1 <<<4, 256, 0, stream>>>(mix_w, wpM);
  feat_k  <<<EW_BLOCKS, 256, 0, stream>>>(x, feat_w, feat_b, xfb);

  for (int t = 0; t < TSTEPS; ++t) {
    // g1 path
    conv1_wmma<<<C1_BLOCKS, 256, 0, stream>>>(h2b, wpG, ur);
    (void)hipMemsetAsync(sums, 0, 64 * sizeof(float), stream);
    stats32<<<512, 256, 0, stream>>>(ur, sums, 0);
    bn_fin<<<1, 32, 0, stream>>>(sums, bn_g + (4 * t + 0) * C, bn_b + (4 * t + 0) * C, 1e-3f, bnA);
    gate1_k<<<EW_BLOCKS, 256, 0, stream>>>(ur, bnA, h2b, hgb);

    // c1 path (heavy)
    conv15_wmma<<<C15_BLOCKS, 256, 0, stream>>>(hgb, wpI, cr);
    (void)hipMemsetAsync(sums, 0, 64 * sizeof(float), stream);
    stats32<<<512, 256, 0, stream>>>(cr, sums, 0);
    bn_fin<<<1, 32, 0, stream>>>(sums, bn_g + (4 * t + 1) * C, bn_b + (4 * t + 1) * C, 1e-3f, bnA);
    h1_k<<<EW_BLOCKS, 256, 0, stream>>>(cr, bnA, xfb, h2b, alpha, mu, h1b);

    // g2 path
    conv1_wmma<<<C1_BLOCKS, 256, 0, stream>>>(h1b, wpM, ur);
    (void)hipMemsetAsync(sums, 0, 64 * sizeof(float), stream);
    stats32<<<512, 256, 0, stream>>>(ur, sums, 0);
    bn_fin<<<1, 32, 0, stream>>>(sums, bn_g + (4 * t + 2) * C, bn_b + (4 * t + 2) * C, 1e-3f, bnA);

    // c2 path (heavy)
    conv15_wmma<<<C15_BLOCKS, 256, 0, stream>>>(h1b, wpE, cr);
    (void)hipMemsetAsync(sums, 0, 64 * sizeof(float), stream);
    stats32<<<512, 256, 0, stream>>>(cr, sums, 0);
    bn_fin<<<1, 32, 0, stream>>>(sums, bn_g + (4 * t + 3) * C, bn_b + (4 * t + 3) * C, 1e-3f, bnB);

    combine_k<<<EW_BLOCKS, 256, 0, stream>>>(ur, bnA, cr, bnB, h1b, h2b, kappa, gamma_p, omega);
  }

  // readout: bn2 -> 1x1 conv to 2ch -> spatial max -> BN1d -> FC
  (void)hipMemsetAsync(sums, 0, 64 * sizeof(float), stream);
  stats32<<<512, 256, 0, stream>>>(h2b, sums, 1);
  bn_fin<<<1, 32, 0, stream>>>(sums, bn2_g, bn2_b, 1e-5f, bnA);
  readout_max_k<<<BATCH * RBLK, 256, 0, stream>>>(h2b, bnA, conv6_w, partial);
  final_k<<<1, 64, 0, stream>>>(partial, bn1_g, bn1_b, fc_w, fc_b, out);
}
